// SimpleStructureModule_16243566313777
// MI455X (gfx1250) — matrix operand order repaired and checked by reading, not compiled, not run on hardware
//
#include <hip/hip_runtime.h>

typedef __attribute__((ext_vector_type(16))) __bf16 v16bf;
typedef __attribute__((ext_vector_type(8)))  __bf16 v8bf;
typedef __attribute__((ext_vector_type(8)))  float  v8f;

union FragAB { v16bf v; v8bf h[2]; };

__device__ __forceinline__ v8f wmma_bf16(v16bf a, v16bf b, v8f c) {
  return __builtin_amdgcn_wmma_f32_16x16x32_bf16(false, a, false, b, (short)0, c, false, false);
}

#define MODE_QKV  0
#define MODE_LN2  1
#define MODE_GELU 2
#define MODE_FFN2 3

// ------------------------------------------------------------------
// Weight convert: W[K][N] fp32 (row-major) -> Wt[N][K] bf16
// ------------------------------------------------------------------
__global__ __launch_bounds__(256)
void wcvt_kernel(const float* __restrict__ W, unsigned short* __restrict__ Wt,
                 int K, int N) {
  int idx = blockIdx.x * 256 + threadIdx.x;
  if (idx >= K * N) return;
  int n = idx / K, k = idx - n * K;
  ((__bf16*)Wt)[idx] = (__bf16)W[(size_t)k * N + n];
}

// ------------------------------------------------------------------
// Pairwise bias: pb[b,h,i,j] = LN(pf[b,i,j,:])@p2h + dist*d2h + causal,
// with src_mask fill (= -1e9). Streams the 512MB tensor once.
// ------------------------------------------------------------------
__global__ __launch_bounds__(256)
void pair_bias_kernel(const float* __restrict__ pfeat, const float* __restrict__ predt,
                      const int* __restrict__ smask, const float* __restrict__ p2h,
                      const float* __restrict__ png, const float* __restrict__ pnb,
                      const float* __restrict__ d2h, float* __restrict__ pb) {
  __shared__ float s_p2h[512], s_g[64], s_b[64], s_d2h[8];
  int t = threadIdx.x;
  for (int i = t; i < 512; i += 256) s_p2h[i] = p2h[i];
  if (t < 64) { s_g[t] = png[t]; s_b[t] = pnb[t]; }
  if (t < 8)  s_d2h[t] = d2h[t];
  __syncthreads();

  size_t gid = (size_t)blockIdx.x * 256 + t;
  int j = (int)(gid & 1023);
  int i = (int)((gid >> 10) & 1023);
  int b = (int)(gid >> 20);

  const float* x = pfeat + gid * 64;
  float4 xs[16];
  #pragma unroll
  for (int q = 0; q < 16; ++q) xs[q] = *(const float4*)(x + q * 4);
  float s1 = 0.f, s2 = 0.f;
  #pragma unroll
  for (int q = 0; q < 16; ++q) {
    s1 += xs[q].x + xs[q].y + xs[q].z + xs[q].w;
    s2 += xs[q].x*xs[q].x + xs[q].y*xs[q].y + xs[q].z*xs[q].z + xs[q].w*xs[q].w;
  }
  float mean = s1 * (1.f/64.f);
  float var  = s2 * (1.f/64.f) - mean * mean;
  float rstd = rsqrtf(var + 1e-5f);

  float acc[8];
  #pragma unroll
  for (int hh = 0; hh < 8; ++hh) acc[hh] = 0.f;
  #pragma unroll
  for (int q = 0; q < 16; ++q) {
    float vv[4] = {xs[q].x, xs[q].y, xs[q].z, xs[q].w};
    #pragma unroll
    for (int c = 0; c < 4; ++c) {
      int p = q * 4 + c;
      float y = (vv[c] - mean) * rstd * s_g[p] + s_b[p];
      #pragma unroll
      for (int hh = 0; hh < 8; ++hh) acc[hh] += y * s_p2h[p * 8 + hh];
    }
  }
  // distance bias
  const float* pj = predt + ((size_t)b * 1024 + j) * 3;
  const float* pi = predt + ((size_t)b * 1024 + i) * 3;
  float dx = pj[0] - pi[0], dy = pj[1] - pi[1], dz = pj[2] - pi[2];
  float d2 = dx*dx + dy*dy + dz*dz;
  d2 = fminf(fmaxf(d2, 2.0f), 1369.0f);
  float dist = sqrtf(d2);
  float cau = (j > i) ? -1e9f : 0.f;
  bool masked = ((smask[b * 1024 + i] == 0) != (smask[b * 1024 + j] == 0));
  #pragma unroll
  for (int hh = 0; hh < 8; ++hh) {
    float val = acc[hh] + dist * s_d2h[hh] + cau;
    if (masked) val = -1e9f;
    pb[(((size_t)b * 8 + hh) * 1024 + i) * 1024 + j] = val;
  }
}

// ------------------------------------------------------------------
// Generic WMMA bf16 GEMM: C[M,N] = X[M,K] (fp32) @ Wt[N,K] (bf16)
// 8 waves/WG, Mtile=32 (2 wave-rows), 256 cols (4 wave-cols x 64).
// Fused epilogues per mode.
// ------------------------------------------------------------------
__global__ __launch_bounds__(256)
void gemm_kernel(const float* __restrict__ X, const unsigned short* __restrict__ Wtu,
                 float* __restrict__ outF, unsigned short* __restrict__ outBu,
                 const float* __restrict__ bias, const float* __restrict__ res,
                 const float* __restrict__ g1, const float* __restrict__ b1, float eps1,
                 const float* __restrict__ g2, const float* __restrict__ b2, float eps2,
                 int K, int N, int Lseq, int mode) {
  __shared__ __attribute__((aligned(16))) __bf16 sA[32 * 40];  // padded stride 40
  __shared__ __attribute__((aligned(16))) float  sC[32 * 256];

  const __bf16* Wt = (const __bf16*)Wtu;
  const int tid = threadIdx.x;
  const int lane = tid & 31, wid = tid >> 5;
  const int wr = wid >> 2, wc = wid & 3;
  const int hf = lane >> 4, l16 = lane & 15;
  const int mblk = blockIdx.x * 32;
  const int nbase = blockIdx.y * 256;

  v8f acc[4] = {};

  for (int k0 = 0; k0 < K; k0 += 32) {
    __syncthreads();
    {  // stage A tile 32x32 fp32 -> bf16 LDS
      int r = tid >> 3, c = (tid & 7) << 2;
      const float4 xv = *(const float4*)(X + (size_t)(mblk + r) * K + k0 + c);
      __bf16* d = &sA[r * 40 + c];
      d[0] = (__bf16)xv.x; d[1] = (__bf16)xv.y; d[2] = (__bf16)xv.z; d[3] = (__bf16)xv.w;
    }
    __syncthreads();
    FragAB a;
    const __bf16* ap = &sA[(wr * 16 + l16) * 40 + hf * 8];
    a.h[0] = *(const v8bf*)ap;
    a.h[1] = *(const v8bf*)(ap + 16);
    #pragma unroll
    for (int f = 0; f < 4; ++f) {
      const __bf16* wp = Wt + (size_t)(nbase + wc * 64 + f * 16 + l16) * K + k0 + hf * 8;
      FragAB bfr;
      bfr.h[0] = *(const v8bf*)wp;
      bfr.h[1] = *(const v8bf*)(wp + 16);
      acc[f] = wmma_bf16(a.v, bfr.v, acc[f]);
    }
  }

  if (mode == MODE_QKV) {
    __bf16* outB = (__bf16*)outBu;
    #pragma unroll
    for (int f = 0; f < 4; ++f) {
      #pragma unroll
      for (int v = 0; v < 8; ++v) {
        int row = mblk + wr * 16 + hf * 8 + v;
        int col = nbase + wc * 64 + f * 16 + l16;
        int bb = row / Lseq, ll = row - bb * Lseq;
        int hh = col >> 5, dd = col & 31;
        outB[(((size_t)bb * 8 + hh) * Lseq + ll) * 32 + dd] = (__bf16)acc[f][v];
      }
    }
    return;
  }
  if (mode == MODE_GELU) {
    #pragma unroll
    for (int f = 0; f < 4; ++f) {
      #pragma unroll
      for (int v = 0; v < 8; ++v) {
        int row = mblk + wr * 16 + hf * 8 + v;
        int col = nbase + wc * 64 + f * 16 + l16;
        float c = acc[f][v] + bias[col];
        outF[(size_t)row * N + col] = 0.5f * c * (1.0f + erff(c * 0.70710678118654752f));
      }
    }
    return;
  }
  // LN modes: dump tile to LDS, then per-row LN with wave reductions
  #pragma unroll
  for (int f = 0; f < 4; ++f)
    #pragma unroll
    for (int v = 0; v < 8; ++v)
      sC[(wr * 16 + hf * 8 + v) * 256 + wc * 64 + f * 16 + l16] = acc[f][v];
  __syncthreads();

  for (int rr = 0; rr < 4; ++rr) {
    int rloc = wid * 4 + rr;
    int grow = mblk + rloc;
    int cbase = lane * 8;
    float x[8];
    float s1 = 0.f, s2 = 0.f;
    #pragma unroll
    for (int i = 0; i < 8; ++i) {
      float c = sC[rloc * 256 + cbase + i];
      if (mode == MODE_FFN2) c += bias[cbase + i];
      c += res[(size_t)grow * 256 + cbase + i];
      x[i] = c; s1 += c; s2 += c * c;
    }
    #pragma unroll
    for (int d = 1; d < 32; d <<= 1) { s1 += __shfl_xor(s1, d, 32); s2 += __shfl_xor(s2, d, 32); }
    float mean = s1 * (1.f/256.f);
    float var  = s2 * (1.f/256.f) - mean * mean;
    float rstd = rsqrtf(var + eps1);
    if (mode == MODE_FFN2) {
      #pragma unroll
      for (int i = 0; i < 8; ++i)
        outF[(size_t)grow * 256 + cbase + i] =
            (x[i] - mean) * rstd * g1[cbase + i] + b1[cbase + i];
    } else {  // MODE_LN2: y = LN(C+res; g1,eps1); z = LN(res + y; g2,eps2)
      float z[8]; float t1 = 0.f, t2 = 0.f;
      #pragma unroll
      for (int i = 0; i < 8; ++i) {
        float y = (x[i] - mean) * rstd * g1[cbase + i] + b1[cbase + i];
        float zz = y + res[(size_t)grow * 256 + cbase + i];
        z[i] = zz; t1 += zz; t2 += zz * zz;
      }
      #pragma unroll
      for (int d = 1; d < 32; d <<= 1) { t1 += __shfl_xor(t1, d, 32); t2 += __shfl_xor(t2, d, 32); }
      float m2 = t1 * (1.f/256.f);
      float v2 = t2 * (1.f/256.f) - m2 * m2;
      float r2 = rsqrtf(v2 + eps2);
      #pragma unroll
      for (int i = 0; i < 8; ++i)
        outF[(size_t)grow * 256 + cbase + i] =
            (z[i] - m2) * r2 * g2[cbase + i] + b2[cbase + i];
    }
  }
}

// ------------------------------------------------------------------
// Flash attention: one WG = (b, h, 64 q rows), 4 waves x 16 rows.
// dk=32 -> Q fragment in registers; K chunk staged via ASYNC global->LDS
// DMA (no VGPR round trip, ASYNCcnt-tracked), V^T staged via VGPRs in
// parallel with the async copies. Bias (pb) carries pairwise + distance
// + causal + mask for self-attention.
// ------------------------------------------------------------------
__global__ __launch_bounds__(128)
void flash_kernel(const unsigned short* __restrict__ Qu, const unsigned short* __restrict__ Ku,
                  const unsigned short* __restrict__ Vu, const float* __restrict__ pbg,
                  float* __restrict__ O, int Lseq, int causal) {
  __shared__ __attribute__((aligned(16))) __bf16 sK[64 * 32];      // [key][d]
  __shared__ __attribute__((aligned(16))) __bf16 sVt[32 * 64];     // [d][key]
  __shared__ __attribute__((aligned(16))) __bf16 sP[4][16 * 64];   // per-wave probs

  const int tid = threadIdx.x, lane = tid & 31, w = tid >> 5;
  const int hf = lane >> 4, l16 = lane & 15;
  const int b = blockIdx.z, h = blockIdx.y;
  const int q0 = blockIdx.x * 64, q0w = q0 + w * 16;
  const size_t bhBase = ((size_t)b * 8 + h) * Lseq;
  const __bf16* Qp = (const __bf16*)Qu + bhBase * 32;
  const __bf16* Kp = (const __bf16*)Ku + bhBase * 32;
  const __bf16* Vp = (const __bf16*)Vu + bhBase * 32;
  const float* pbp = pbg ? (pbg + bhBase * Lseq) : (const float*)0;

  FragAB qf;
  {
    const __bf16* p = Qp + (size_t)(q0w + l16) * 32 + hf * 8;
    qf.h[0] = *(const v8bf*)p;
    qf.h[1] = *(const v8bf*)(p + 16);
  }

  v8f o0 = {}, o1 = {};
  float Mrow[8], Lsum[8];
  #pragma unroll
  for (int v = 0; v < 8; ++v) { Mrow[v] = -1e30f; Lsum[v] = 0.f; }

  const float scl = 0.17677669529663687f;  // 1/sqrt(32)
  const int jend = causal ? (q0 + 64) : Lseq;

  // per-thread staging slice: 32B of K row n, 32B of V row n
  const int sn = tid >> 1, sd0 = (tid & 1) * 16;
  const unsigned int ldsK = (unsigned int)(size_t)(&sK[sn * 32 + sd0]);

  for (int j0 = 0; j0 < jend; j0 += 64) {
    __syncthreads();
    {
      // K tile: pure memcpy -> async global->LDS DMA (INST_OFFSET applies
      // to both the global and LDS sides, so one LDS-addr VGPR serves both
      // 16B beats). Tracked by ASYNCcnt.
      const __bf16* kp = Kp + (size_t)(j0 + sn) * 32 + sd0;
      asm volatile(
          "global_load_async_to_lds_b128 %0, %1, off\n\t"
          "global_load_async_to_lds_b128 %0, %1, off offset:16"
          :: "v"(ldsK), "v"(kp) : "memory");
      // V tile needs a transpose -> regular path, overlaps the async copy.
      const __bf16* vp = Vp + (size_t)(j0 + sn) * 32 + sd0;
      v8bf va = *(const v8bf*)vp;
      v8bf vb = *(const v8bf*)(vp + 8);
      #pragma unroll
      for (int i = 0; i < 8; ++i) sVt[(sd0 + i) * 64 + sn] = va[i];
      #pragma unroll
      for (int i = 0; i < 8; ++i) sVt[(sd0 + 8 + i) * 64 + sn] = vb[i];
      if (j0 + 64 < jend) {
        __builtin_prefetch(Vp + (size_t)(j0 + 64) * 32, 0, 1);
      }
    }
    // our async writes must land before we signal the barrier
    asm volatile("s_wait_asynccnt 0x0" ::: "memory");
    __syncthreads();

    // S = Q K^T
    v8f s[4];
    #pragma unroll
    for (int kt = 0; kt < 4; ++kt) {
      FragAB kb;
      const __bf16* p = &sK[(kt * 16 + l16) * 32 + hf * 8];
      kb.h[0] = *(const v8bf*)p;
      kb.h[1] = *(const v8bf*)(p + 16);
      v8f z = {};
      s[kt] = wmma_bf16(qf.v, kb.v, z);
    }
    // scale + bias
    #pragma unroll
    for (int kt = 0; kt < 4; ++kt) {
      #pragma unroll
      for (int v = 0; v < 8; ++v) {
        float val = s[kt][v] * scl;
        if (pbp) {
          int qrow = q0w + hf * 8 + v;
          val += pbp[(size_t)qrow * Lseq + j0 + kt * 16 + l16];
        }
        s[kt][v] = val;
      }
    }
    // online softmax (row stats via intra-half shfl reductions)
    #pragma unroll
    for (int v = 0; v < 8; ++v) {
      float m = fmaxf(fmaxf(s[0][v], s[1][v]), fmaxf(s[2][v], s[3][v]));
      #pragma unroll
      for (int d = 1; d < 16; d <<= 1) m = fmaxf(m, __shfl_xor(m, d, 32));
      float mn = fmaxf(Mrow[v], m);
      float sc = __expf(Mrow[v] - mn);
      Mrow[v] = mn;
      float ps = 0.f;
      #pragma unroll
      for (int kt = 0; kt < 4; ++kt) {
        float p = __expf(s[kt][v] - mn);
        s[kt][v] = p; ps += p;
      }
      #pragma unroll
      for (int d = 1; d < 16; d <<= 1) ps += __shfl_xor(ps, d, 32);
      Lsum[v] = Lsum[v] * sc + ps;
      o0[v] *= sc; o1[v] *= sc;
      int r = hf * 8 + v;
      #pragma unroll
      for (int kt = 0; kt < 4; ++kt)
        sP[w][r * 64 + kt * 16 + l16] = (__bf16)s[kt][v];
    }
    asm volatile("s_wait_dscnt 0" ::: "memory");
    // O += P V
    #pragma unroll
    for (int ks = 0; ks < 2; ++ks) {
      FragAB pf;
      const __bf16* pp = &sP[w][l16 * 64 + ks * 32 + hf * 8];
      pf.h[0] = *(const v8bf*)pp;
      pf.h[1] = *(const v8bf*)(pp + 16);
      #pragma unroll
      for (int dt = 0; dt < 2; ++dt) {
        FragAB vf;
        const __bf16* vp = &sVt[(dt * 16 + l16) * 64 + ks * 32 + hf * 8];
        vf.h[0] = *(const v8bf*)vp;
        vf.h[1] = *(const v8bf*)(vp + 16);
        if (dt == 0) o0 = wmma_bf16(pf.v, vf.v, o0);
        else         o1 = wmma_bf16(pf.v, vf.v, o1);
      }
    }
  }
  // normalize + write [b, l, h*32+d]
  #pragma unroll
  for (int v = 0; v < 8; ++v) {
    float inv = 1.f / Lsum[v];
    int q = q0w + hf * 8 + v;
    float* op = O + ((size_t)b * Lseq + q) * 256 + h * 32;
    op[l16]      = o0[v] * inv;
    op[16 + l16] = o1[v] * inv;
  }
}

// ------------------------------------------------------------------
extern "C" void kernel_launch(void* const* d_in, const int* in_sizes, int n_in,
                              void* d_out, int out_size, void* d_ws, size_t ws_size,
                              hipStream_t stream) {
  (void)in_sizes; (void)n_in; (void)out_size; (void)ws_size;
  const int B = 2, L = 1024, D = 256, H = 8, F = 1024, M = B * L;

  const float* tgt   = (const float*)d_in[0];
  const float* src   = (const float*)d_in[1];
  const float* pf    = (const float*)d_in[2];
  const float* predt = (const float*)d_in[3];
  const int*   smask = (const int*)d_in[4];
  const float* sa_wq = (const float*)d_in[5];
  const float* sa_wk = (const float*)d_in[6];
  const float* sa_wv = (const float*)d_in[7];
  const float* sa_wo = (const float*)d_in[8];
  const float* sa_g  = (const float*)d_in[9];
  const float* sa_b  = (const float*)d_in[10];
  const float* ca_wq = (const float*)d_in[11];
  const float* ca_wk = (const float*)d_in[12];
  const float* ca_wv = (const float*)d_in[13];
  const float* ca_wo = (const float*)d_in[14];
  const float* ca_g  = (const float*)d_in[15];
  const float* ca_b  = (const float*)d_in[16];
  const float* p2h   = (const float*)d_in[17];
  const float* pn_g  = (const float*)d_in[18];
  const float* pn_b  = (const float*)d_in[19];
  const float* d2h   = (const float*)d_in[20];
  const float* l1w   = (const float*)d_in[21];
  const float* l1b   = (const float*)d_in[22];
  const float* l2w   = (const float*)d_in[23];
  const float* l2b   = (const float*)d_in[24];
  const float* n1g   = (const float*)d_in[25];
  const float* n1b   = (const float*)d_in[26];
  const float* n2g   = (const float*)d_in[27];
  const float* n2b   = (const float*)d_in[28];
  const float* n3g   = (const float*)d_in[29];
  const float* n3b   = (const float*)d_in[30];
  float* out = (float*)d_out;

  char* ws = (char*)d_ws;
  size_t off = 0;
  auto alloc = [&](size_t bytes) -> void* {
    void* p = ws + off;
    off += (bytes + 255) & ~(size_t)255;
    return p;
  };
  // ~84 MB total workspace
  float* pb = (float*)alloc((size_t)B * H * L * L * sizeof(float));     // 64 MB
  unsigned short* wQ1 = (unsigned short*)alloc((size_t)D * D * 2);
  unsigned short* wK1 = (unsigned short*)alloc((size_t)D * D * 2);
  unsigned short* wV1 = (unsigned short*)alloc((size_t)D * D * 2);
  unsigned short* wO1 = (unsigned short*)alloc((size_t)D * D * 2);
  unsigned short* wQ2 = (unsigned short*)alloc((size_t)D * D * 2);
  unsigned short* wK2 = (unsigned short*)alloc((size_t)D * D * 2);
  unsigned short* wV2 = (unsigned short*)alloc((size_t)D * D * 2);
  unsigned short* wO2 = (unsigned short*)alloc((size_t)D * D * 2);
  unsigned short* wl1 = (unsigned short*)alloc((size_t)D * F * 2);
  unsigned short* wl2 = (unsigned short*)alloc((size_t)F * D * 2);
  unsigned short* Qb  = (unsigned short*)alloc((size_t)M * D * 2);
  unsigned short* Kb  = (unsigned short*)alloc((size_t)M * D * 2);
  unsigned short* Vb  = (unsigned short*)alloc((size_t)M * D * 2);
  float* AO = (float*)alloc((size_t)M * D * sizeof(float));
  float* T1 = (float*)alloc((size_t)M * D * sizeof(float));
  float* U  = (float*)alloc((size_t)M * D * sizeof(float));
  float* Hd = (float*)alloc((size_t)M * F * sizeof(float));

  auto wc = [&](const float* W, unsigned short* Wt, int K, int N) {
    wcvt_kernel<<<dim3((K * N + 255) / 256), 256, 0, stream>>>(W, Wt, K, N);
  };
  auto gemm = [&](const float* X, const unsigned short* Wt, float* oF, unsigned short* oB,
                  const float* bias, const float* res,
                  const float* g1, const float* b1, float e1,
                  const float* g2, const float* b2, float e2,
                  int K, int N, int mode) {
    gemm_kernel<<<dim3(M / 32, N / 256), 256, 0, stream>>>(
        X, Wt, oF, oB, bias, res, g1, b1, e1, g2, b2, e2, K, N, L, mode);
  };

  // 1. weights -> bf16 transposed
  wc(sa_wq, wQ1, D, D); wc(sa_wk, wK1, D, D); wc(sa_wv, wV1, D, D); wc(sa_wo, wO1, D, D);
  wc(ca_wq, wQ2, D, D); wc(ca_wk, wK2, D, D); wc(ca_wv, wV2, D, D); wc(ca_wo, wO2, D, D);
  wc(l1w, wl1, D, F);   wc(l2w, wl2, F, D);

  // 2. pairwise + distance + causal + mask bias
  pair_bias_kernel<<<dim3((B * L * L) / 256), 256, 0, stream>>>(
      pf, predt, smask, p2h, pn_g, pn_b, d2h, pb);

  // 3. self-attention QKV projections (bf16, [b,h,l,d])
  gemm(tgt, wQ1, 0, Qb, 0, 0, 0, 0, 0.f, 0, 0, 0.f, D, D, MODE_QKV);
  gemm(tgt, wK1, 0, Kb, 0, 0, 0, 0, 0.f, 0, 0, 0.f, D, D, MODE_QKV);
  gemm(tgt, wV1, 0, Vb, 0, 0, 0, 0, 0.f, 0, 0, 0.f, D, D, MODE_QKV);

  // 4. self-attention (biased, causal kv-loop limit)
  flash_kernel<<<dim3(L / 64, H, B), 128, 0, stream>>>(Qb, Kb, Vb, pb, AO, L, 1);

  // 5. out-proj + LN(sa_ln,1e-6) + LN(n1,1e-5), residual = tgt
  gemm(AO, wO1, T1, 0, 0, tgt, sa_g, sa_b, 1e-6f, n1g, n1b, 1e-5f, D, D, MODE_LN2);

  // 6. cross-attention QKV
  gemm(T1, wQ2, 0, Qb, 0, 0, 0, 0, 0.f, 0, 0, 0.f, D, D, MODE_QKV);
  gemm(src, wK2, 0, Kb, 0, 0, 0, 0, 0.f, 0, 0, 0.f, D, D, MODE_QKV);
  gemm(src, wV2, 0, Vb, 0, 0, 0, 0, 0.f, 0, 0, 0.f, D, D, MODE_QKV);

  // 7. cross-attention (no bias)
  flash_kernel<<<dim3(L / 64, H, B), 128, 0, stream>>>(Qb, Kb, Vb, (const float*)0, AO, L, 0);

  // 8. out-proj + LN(ca_ln,1e-6) + LN(n3,1e-5), residual = T1
  gemm(AO, wO2, U, 0, 0, T1, ca_g, ca_b, 1e-6f, n3g, n3b, 1e-5f, D, D, MODE_LN2);

  // 9. FFN up + exact GELU
  gemm(U, wl1, Hd, 0, l1b, 0, 0, 0, 0.f, 0, 0, 0.f, D, F, MODE_GELU);

  // 10. FFN down + bias + residual + LN(n2,1e-5) -> d_out
  gemm(Hd, wl2, out, 0, l2b, U, n2g, n2b, 1e-5f, 0, 0, 0.f, F, D, MODE_FFN2);
}
